// SingleHeadAttention_12025908429074
// MI455X (gfx1250) — compile-verified, matrix operand order checked by reading, not run on hardware
//
#include <hip/hip_runtime.h>
#include <hip/hip_bf16.h>

// ---------------------------------------------------------------------------
// Single-head causal attention for MI455X (gfx1250, wave32, WMMA bf16).
// B=4, T=2048, C=H=1024.  Pipeline:
//   1) proj_gemm:       kp=k*Wk, qp=q*Wq  (fp32 -> bf16, row-major)
//      proj_gemm<TR>:   vpT=(v*Wq)^T      (fp32 -> bf16, stored [B][H][T])
//   2) scores:  S = (qp . kp^T) * H^-0.5  (async global->LDS double-buffered,
//               bf16 WMMA, fp32 out; upper causal tiles skipped; S fits in L2)
//   3) softmax: row softmax over s<=t, P bf16 (zeros above diagonal),
//               P reuses the qp/kp workspace region
//   4) out_gemm: O = P . vpT^T (async double-buffered, bf16 WMMA, fp32 out),
//               K-loop stops at t0+128 (causal zeros in P)
// Workspace layout (needs >= 112 MiB):
//   [0,16M)   qp bf16   -> later reused as P (32 MiB)
//   [16,32M)  kp bf16
//   [32,48M)  vpT bf16  [B][H][T]
//   [48,112M) S  fp32
// ---------------------------------------------------------------------------

#define LDK 40   // LDS row stride in ushorts (80B = 5*16B: aligned, bank-spread)

typedef __attribute__((ext_vector_type(16))) __bf16 v16bf;
typedef __attribute__((ext_vector_type(8)))  float  v8f;
typedef unsigned long long u64;

union FragU { uint4 u4[2]; v16bf v; };

__device__ __forceinline__ unsigned short f32_to_bf16(float f) {
  unsigned int u = __float_as_uint(f);
  u += 0x7FFFu + ((u >> 16) & 1u);          // round-to-nearest-even
  return (unsigned short)(u >> 16);
}

// 16x32 bf16 operand fragment per ISA layout:
// lanes 0-15: row=lane,   K={0..7,16..23}; lanes 16-31: row=lane-16, K={8..15,24..31}
__device__ __forceinline__ v16bf load_frag(const unsigned short* base) {
  const int lane = threadIdx.x & 31;
  const unsigned short* p = base + (lane & 15) * LDK + ((lane >> 4) << 3);
  FragU f;
  f.u4[0] = *(const uint4*)(p);
  f.u4[1] = *(const uint4*)(p + 16);
  return f.v;
}

__device__ __forceinline__ v8f wmma_bf16(v16bf a, v16bf b, v8f c) {
  return __builtin_amdgcn_wmma_f32_16x16x32_bf16(
      /*neg_a=*/false, a, /*neg_b=*/false, b,
      /*c_mod=*/(short)0, c, /*reuse_a=*/false, /*reuse_b=*/false);
}

// Async global->LDS copy of a 128x32 bf16 tile into a padded [128][LDK] LDS
// tile. 512 b128 transfers, 2 per thread, tracked by ASYNCcnt. LDS flat
// addresses carry the LDS byte offset in addr[31:0] (ISA 10.2), so a 32-bit
// truncation of the generic pointer is the async vdst address.
__device__ __forceinline__ void async_copy_tile(unsigned short* lds_tile,
                                                const unsigned short* gbase,
                                                size_t g_ld) {
  const int tid = threadIdx.x;
#pragma unroll
  for (int i = 0; i < 2; ++i) {
    const int idx = tid + (i << 8);          // 0..511 b128 slots
    const int row = idx >> 2;                // 0..127
    const int c8  = (idx & 3) << 3;          // 0,8,16,24 ushorts (16B chunks)
    const unsigned lds_addr = (unsigned)(size_t)(lds_tile + row * LDK + c8);
    const u64 gaddr = (u64)(size_t)(gbase + (size_t)row * g_ld + c8);
    asm volatile("global_load_async_to_lds_b128 %0, %1, off"
                 :: "v"(lds_addr), "v"(gaddr) : "memory");
  }
}

__device__ __forceinline__ void wait_async_le4() {
  asm volatile("s_wait_asynccnt 0x4" ::: "memory");
}
__device__ __forceinline__ void wait_async_0() {
  asm volatile("s_wait_asynccnt 0x0" ::: "memory");
}

// ---------------------------------------------------------------------------
// Kernel 1: A[8192x1024](f32) x W[1024x1024](f32) -> bf16
//   TR=false: Out[m][n], ld 1024        (qp, kp)
//   TR=true : Out[b][n][t], ld 2048     (vpT, b=m>>11, t=m&2047)
// ---------------------------------------------------------------------------
template <bool TR>
__global__ __launch_bounds__(256)
void proj_gemm_kernel(const float* __restrict__ A, const float* __restrict__ W,
                      unsigned short* __restrict__ Out) {
  __shared__ __align__(16) unsigned short As[128 * LDK];
  __shared__ __align__(16) unsigned short Bs[128 * LDK];
  const int tid = threadIdx.x;
  const int m0  = blockIdx.y * 128;
  const int n0  = blockIdx.x * 128;
  const int w   = tid >> 5;
  const int wm  = (w >> 1) * 32;
  const int wn  = (w & 1) * 64;

  const v8f vzero = {0.f,0.f,0.f,0.f,0.f,0.f,0.f,0.f};
  v8f acc[2][4];
#pragma unroll
  for (int i = 0; i < 2; ++i)
#pragma unroll
    for (int j = 0; j < 4; ++j) acc[i][j] = vzero;

  for (int k0 = 0; k0 < 1024; k0 += 32) {
    // stage A tile 128x32, fp32 -> bf16
#pragma unroll
    for (int i = 0; i < 4; ++i) {
      const int idx = tid + (i << 8);          // 1024 float4 slots
      const int row = idx >> 3;
      const int kk  = (idx & 7) << 2;
      const float4 a = *(const float4*)(A + (size_t)(m0 + row) * 1024 + k0 + kk);
      unsigned short* d = &As[row * LDK + kk];
      d[0] = f32_to_bf16(a.x); d[1] = f32_to_bf16(a.y);
      d[2] = f32_to_bf16(a.z); d[3] = f32_to_bf16(a.w);
    }
    // stage B tile transposed: Bs[n][k] = W[k0+k][n0+n]
#pragma unroll
    for (int i = 0; i < 4; ++i) {
      const int idx = tid + (i << 8);
      const int kk  = idx >> 5;                // 0..31
      const int n4  = (idx & 31) << 2;         // 0..124
      const float4 wv = *(const float4*)(W + (size_t)(k0 + kk) * 1024 + n0 + n4);
      Bs[(n4 + 0) * LDK + kk] = f32_to_bf16(wv.x);
      Bs[(n4 + 1) * LDK + kk] = f32_to_bf16(wv.y);
      Bs[(n4 + 2) * LDK + kk] = f32_to_bf16(wv.z);
      Bs[(n4 + 3) * LDK + kk] = f32_to_bf16(wv.w);
    }
    if (k0 + 32 < 1024)                        // global_prefetch_b8 next A tile
      __builtin_prefetch(A + (size_t)(m0 + (tid & 127)) * 1024 + k0 + 32, 0, 1);
    __syncthreads();

    v16bf af[2], bfr[4];
    af[0] = load_frag(&As[(wm     ) * LDK]);
    af[1] = load_frag(&As[(wm + 16) * LDK]);
#pragma unroll
    for (int j = 0; j < 4; ++j) bfr[j] = load_frag(&Bs[(wn + 16 * j) * LDK]);
#pragma unroll
    for (int i = 0; i < 2; ++i)
#pragma unroll
      for (int j = 0; j < 4; ++j) acc[i][j] = wmma_bf16(af[i], bfr[j], acc[i][j]);
    __syncthreads();
  }

  const int lane = tid & 31;
  const int half = lane >> 4;
  const int nl   = lane & 15;
#pragma unroll
  for (int i = 0; i < 2; ++i)
#pragma unroll
    for (int j = 0; j < 4; ++j)
#pragma unroll
      for (int r = 0; r < 8; ++r) {
        const int m = m0 + wm + 16 * i + r + 8 * half;
        const int n = n0 + wn + 16 * j + nl;
        const unsigned short o = f32_to_bf16(acc[i][j][r]);
        if (TR) Out[((size_t)(m >> 11) * 1024 + n) * 2048 + (m & 2047)] = o;
        else    Out[(size_t)m * 1024 + n] = o;
      }
}

// ---------------------------------------------------------------------------
// Kernel 2: S[b][t][s] = scale * sum_h qp[b,t,h]*kp[b,s,h]  (lower tiles only)
// Async double-buffered staging; both operands use A-style fragments.
// ---------------------------------------------------------------------------
__global__ __launch_bounds__(256)
void scores_kernel(const unsigned short* __restrict__ Qp,
                   const unsigned short* __restrict__ Kp,
                   float* __restrict__ S) {
  const int st = blockIdx.x;                   // key tile
  const int tt = blockIdx.y;                   // query tile
  if (st > tt) return;                         // fully above diagonal
  const int b  = blockIdx.z;
  const int t0 = tt * 128, s0 = st * 128;
  const unsigned short* Q = Qp + (size_t)b * 2048 * 1024 + (size_t)t0 * 1024;
  const unsigned short* K = Kp + (size_t)b * 2048 * 1024 + (size_t)s0 * 1024;
  float* Sb = S + (size_t)b * 2048 * 2048;

  __shared__ __align__(16) unsigned short As[2][128 * LDK];
  __shared__ __align__(16) unsigned short Bs[2][128 * LDK];
  const int tid = threadIdx.x;
  const int w   = tid >> 5;
  const int wm  = (w >> 1) * 32;
  const int wn  = (w & 1) * 64;

  const v8f vzero = {0.f,0.f,0.f,0.f,0.f,0.f,0.f,0.f};
  v8f acc[2][4];
#pragma unroll
  for (int i = 0; i < 2; ++i)
#pragma unroll
    for (int j = 0; j < 4; ++j) acc[i][j] = vzero;

  async_copy_tile(As[0], Q, 1024);             // prologue: tile 0 in flight
  async_copy_tile(Bs[0], K, 1024);

  for (int it = 0; it < 32; ++it) {
    const int k0 = it * 32;
    if (it + 1 < 32) {                         // stream tile it+1 behind
      async_copy_tile(As[(it + 1) & 1], Q + k0 + 32, 1024);
      async_copy_tile(Bs[(it + 1) & 1], K + k0 + 32, 1024);
      wait_async_le4();                        // retire tile it's 4 transfers
    } else {
      wait_async_0();
    }
    __syncthreads();

    const unsigned short* a = As[it & 1];
    const unsigned short* bb = Bs[it & 1];
    v16bf af[2], bfr[4];
    af[0] = load_frag(&a[(wm     ) * LDK]);
    af[1] = load_frag(&a[(wm + 16) * LDK]);
#pragma unroll
    for (int j = 0; j < 4; ++j) bfr[j] = load_frag(&bb[(wn + 16 * j) * LDK]);
#pragma unroll
    for (int i = 0; i < 2; ++i)
#pragma unroll
      for (int j = 0; j < 4; ++j) acc[i][j] = wmma_bf16(af[i], bfr[j], acc[i][j]);
    __syncthreads();
  }

  const float scale = 0.03125f;                // 1024^-0.5
  const int lane = tid & 31;
  const int half = lane >> 4;
  const int nl   = lane & 15;
#pragma unroll
  for (int i = 0; i < 2; ++i)
#pragma unroll
    for (int j = 0; j < 4; ++j)
#pragma unroll
      for (int r = 0; r < 8; ++r) {
        const int t = t0 + wm + 16 * i + r + 8 * half;
        const int s = s0 + wn + 16 * j + nl;
        Sb[(size_t)t * 2048 + s] = acc[i][j][r] * scale;  // s>t garbage never read
      }
}

// ---------------------------------------------------------------------------
// Kernel 3: per-row causal softmax; P bf16 (zero above diagonal)
// ---------------------------------------------------------------------------
__global__ __launch_bounds__(256)
void softmax_kernel(const float* __restrict__ S, unsigned short* __restrict__ P) {
  const int row = blockIdx.x;                  // b*2048 + t
  const int t   = row & 2047;
  const float* Sr = S + (size_t)row * 2048;
  unsigned short* Pr = P + (size_t)row * 2048;
  const int tid = threadIdx.x;

  float vals[8];
#pragma unroll
  for (int i = 0; i < 8; ++i) {
    const int s = tid + (i << 8);
    vals[i] = (s <= t) ? Sr[s] : -3.0e38f;
  }
  float m = vals[0];
#pragma unroll
  for (int i = 1; i < 8; ++i) m = fmaxf(m, vals[i]);

  __shared__ float red[256];
  red[tid] = m; __syncthreads();
  for (int off = 128; off > 0; off >>= 1) {
    if (tid < off) red[tid] = fmaxf(red[tid], red[tid + off]);
    __syncthreads();
  }
  m = red[0]; __syncthreads();

  float e[8]; float sum = 0.f;
#pragma unroll
  for (int i = 0; i < 8; ++i) {
    const int s = tid + (i << 8);
    e[i] = (s <= t) ? __expf(vals[i] - m) : 0.f;
    sum += e[i];
  }
  red[tid] = sum; __syncthreads();
  for (int off = 128; off > 0; off >>= 1) {
    if (tid < off) red[tid] += red[tid + off];
    __syncthreads();
  }
  const float inv = 1.0f / red[0];
#pragma unroll
  for (int i = 0; i < 8; ++i) {
    const int s = tid + (i << 8);
    Pr[s] = (s <= t) ? f32_to_bf16(e[i] * inv) : (unsigned short)0;
  }
}

// ---------------------------------------------------------------------------
// Kernel 4: O[b][t][h] = sum_s P[b,t,s]*vpT[b,h,s]
// Async double-buffered staging; K-loop truncated at t0+128 (causal zeros).
// ---------------------------------------------------------------------------
__global__ __launch_bounds__(256)
void out_gemm_kernel(const unsigned short* __restrict__ P,
                     const unsigned short* __restrict__ Vt,
                     float* __restrict__ O) {
  const int b  = blockIdx.z;
  const int t0 = blockIdx.y * 128;
  const int h0 = blockIdx.x * 128;
  const unsigned short* Pb = P  + (size_t)b * 2048 * 2048 + (size_t)t0 * 2048;
  const unsigned short* Vb = Vt + ((size_t)b * 1024 + h0) * 2048;  // [h][t]
  float* Ob = O + (size_t)b * 2048 * 1024;

  __shared__ __align__(16) unsigned short As[2][128 * LDK];
  __shared__ __align__(16) unsigned short Bs[2][128 * LDK];
  const int tid = threadIdx.x;
  const int w   = tid >> 5;
  const int wm  = (w >> 1) * 32;
  const int wn  = (w & 1) * 64;

  const v8f vzero = {0.f,0.f,0.f,0.f,0.f,0.f,0.f,0.f};
  v8f acc[2][4];
#pragma unroll
  for (int i = 0; i < 2; ++i)
#pragma unroll
    for (int j = 0; j < 4; ++j) acc[i][j] = vzero;

  const int nIt = (t0 + 128) >> 5;             // P rows are zero beyond t
  async_copy_tile(As[0], Pb, 2048);
  async_copy_tile(Bs[0], Vb, 2048);

  for (int it = 0; it < nIt; ++it) {
    const int k0 = it * 32;
    if (it + 1 < nIt) {
      async_copy_tile(As[(it + 1) & 1], Pb + k0 + 32, 2048);
      async_copy_tile(Bs[(it + 1) & 1], Vb + k0 + 32, 2048);
      wait_async_le4();
    } else {
      wait_async_0();
    }
    __syncthreads();

    const unsigned short* a = As[it & 1];
    const unsigned short* bb = Bs[it & 1];
    v16bf af[2], bfr[4];
    af[0] = load_frag(&a[(wm     ) * LDK]);
    af[1] = load_frag(&a[(wm + 16) * LDK]);
#pragma unroll
    for (int j = 0; j < 4; ++j) bfr[j] = load_frag(&bb[(wn + 16 * j) * LDK]);
#pragma unroll
    for (int i = 0; i < 2; ++i)
#pragma unroll
      for (int j = 0; j < 4; ++j) acc[i][j] = wmma_bf16(af[i], bfr[j], acc[i][j]);
    __syncthreads();
  }

  const int lane = tid & 31;
  const int half = lane >> 4;
  const int nl   = lane & 15;
#pragma unroll
  for (int i = 0; i < 2; ++i)
#pragma unroll
    for (int j = 0; j < 4; ++j)
#pragma unroll
      for (int r = 0; r < 8; ++r) {
        const int t = t0 + wm + 16 * i + r + 8 * half;
        const int h = h0 + wn + 16 * j + nl;
        Ob[(size_t)t * 1024 + h] = acc[i][j][r];
      }
}

// ---------------------------------------------------------------------------
extern "C" void kernel_launch(void* const* d_in, const int* in_sizes, int n_in,
                              void* d_out, int out_size, void* d_ws, size_t ws_size,
                              hipStream_t stream) {
  (void)in_sizes; (void)n_in; (void)out_size; (void)ws_size;
  // inputs: x, k, q, v, Wk, Wq, Wv  (x and Wv unused, as in reference)
  const float* k  = (const float*)d_in[1];
  const float* q  = (const float*)d_in[2];
  const float* v  = (const float*)d_in[3];
  const float* Wk = (const float*)d_in[4];
  const float* Wq = (const float*)d_in[5];

  unsigned short* qp  = (unsigned short*)d_ws;                 // 16 MiB
  unsigned short* kp  = qp + (size_t)8192 * 1024;              // 16 MiB
  unsigned short* vpT = kp + (size_t)8192 * 1024;              // 16 MiB, [B][H][T]
  float* S = (float*)((char*)d_ws + (size_t)48 * 1024 * 1024); // 64 MiB
  unsigned short* P = (unsigned short*)d_ws;                   // reuse qp+kp

  dim3 blk(256);
  dim3 gproj(8, 64);
  proj_gemm_kernel<false><<<gproj, blk, 0, stream>>>(q, Wq, qp);
  proj_gemm_kernel<false><<<gproj, blk, 0, stream>>>(k, Wk, kp);
  proj_gemm_kernel<true ><<<gproj, blk, 0, stream>>>(v, Wq, vpT); // faithful: v uses Wq

  scores_kernel <<<dim3(16, 16, 4), blk, 0, stream>>>(qp, kp, S);
  softmax_kernel<<<dim3(8192),      blk, 0, stream>>>(S, P);
  out_gemm_kernel<<<dim3(8, 16, 4), blk, 0, stream>>>(P, vpT, (float*)d_out);
}
